// OneScalePredictor_45801531245056
// MI455X (gfx1250) — matrix operand order repaired and checked by reading, not compiled, not run on hardware
//
#include <hip/hip_runtime.h>

typedef __attribute__((ext_vector_type(16))) _Float16 v16h;
typedef __attribute__((ext_vector_type(8)))  _Float16 v8h;
typedef __attribute__((ext_vector_type(8)))  float    v8f;
typedef __attribute__((ext_vector_type(4)))  int      v4i;

#define TAPS 27
#define CH   32

// ---------------------------------------------------------------------------
// Init conv: rec_F (N,1) -> F0 (N,32) f16.  out[p,c] = b[c] + sum_k W[k,0,c]*g
// ---------------------------------------------------------------------------
__global__ void init_conv_kernel(const float* __restrict__ recF,
                                 const float* __restrict__ Wi,   // (27,1,32)
                                 const float* __restrict__ bi,   // (32)
                                 const int*   __restrict__ nbr,  // (27,N)
                                 _Float16* __restrict__ out, int n) {
    int p = blockIdx.x * 256 + threadIdx.x;
    if (p >= n) return;
    float acc[CH];
#pragma unroll
    for (int c = 0; c < CH; ++c) acc[c] = bi[c];
    const int* nbp = nbr + p;
    for (int k = 0; k < TAPS; ++k) {
        int idx = *nbp; nbp += n;
        float g = 0.0f;
        if (idx >= 0) g = recF[idx];
#pragma unroll
        for (int c = 0; c < CH; ++c) acc[c] = fmaf(g, Wi[k * CH + c], acc[c]);
    }
#pragma unroll
    for (int j = 0; j < 4; ++j) {
        v8h ch;
#pragma unroll
        for (int e = 0; e < 8; ++e) ch[e] = (_Float16)acc[j * 8 + e];
        *(v8h*)(out + (size_t)p * CH + j * 8) = ch;
    }
}

// ---------------------------------------------------------------------------
// Sparse 27-tap conv 32->32, fused bias (+optional residual) + PReLU.
// Block = 256 thr = 8 waves; wave owns 16 points; 27 taps x 2 Cout tiles
// = 54 x v_wmma_f32_16x16x32_f16 (f32 accumulate). Branchless masked gather.
// ---------------------------------------------------------------------------
__global__ void sparse_conv_wmma(const _Float16* __restrict__ Fin,   // (N,32) f16
                                 const float*    __restrict__ Wg,    // (27,32,32) [k][cin][cout]
                                 const float*    __restrict__ bias,  // (32)
                                 const float*    __restrict__ alphaP,// scalar
                                 const _Float16* __restrict__ res,   // (N,32) f16 or null
                                 const int*      __restrict__ nbr,   // (27,N)
                                 _Float16* __restrict__ Fout, int n) {
    // Weights transposed to [k][cout][cin] f16 in LDS (55296 B): B fragment is
    // one contiguous 32B LDS read per lane, k = 16*half + s, same order as A.
    __shared__ v16h WldsV[TAPS * CH * CH / 16];
    _Float16* Wlds = (_Float16*)WldsV;
    for (int i = threadIdx.x; i < TAPS * CH * CH; i += 256) {
        int k = i >> 10, r = i & 1023, co = r >> 5, ci = r & 31;
        Wlds[i] = (_Float16)Wg[k * 1024 + ci * CH + co];
    }
    __syncthreads();

    const int lane  = threadIdx.x & 31;
    const int wave  = threadIdx.x >> 5;
    const int m_base = blockIdx.x * 128 + wave * 16;
    const int lr    = lane & 15;
    const bool lo16 = lane < 16;
    const int kbA   = lo16 ? 0 : 16;   // A: lanes<16 hold K0-15
    const int kbB   = lo16 ? 0 : 16;   // B: lanes<16 hold K0-15 of col N=lane

    int prow  = m_base + lr;
    int prowc = prow < n ? prow : n - 1;
    const int* nbp = nbr + prowc;

    v8f c0 = {0.f,0.f,0.f,0.f,0.f,0.f,0.f,0.f};
    v8f c1 = {0.f,0.f,0.f,0.f,0.f,0.f,0.f,0.f};

#pragma unroll 3
    for (int k = 0; k < TAPS; ++k) {
        int idx = *nbp; nbp += n;
        // Branchless masked gather: clamp index, load, AND with sign mask.
        int idxc = idx < 0 ? 0 : idx;                  // v_max_i32
        int keep = ~(idx >> 31);                       // 0xFFFFFFFF if valid
        const _Float16* p = Fin + (size_t)idxc * CH + kbA;
        v4i lo = *(const v4i*)(p);
        v4i hi = *(const v4i*)(p + 8);
        v4i kv = {keep, keep, keep, keep};
        lo &= kv; hi &= kv;
        union { v16h h; v4i i[2]; } au;
        au.i[0] = lo; au.i[1] = hi;
        v16h a = au.h;

        v16h b0 = *(const v16h*)(Wlds + ((k * CH +  0 + lr) * CH + kbB));
        v16h b1 = *(const v16h*)(Wlds + ((k * CH + 16 + lr) * CH + kbB));

        c0 = __builtin_amdgcn_wmma_f32_16x16x32_f16(false, a, false, b0,
                                                    (short)0, c0, false, false);
        c1 = __builtin_amdgcn_wmma_f32_16x16x32_f16(false, a, false, b1,
                                                    (short)0, c1, false, false);
    }

    const float alpha = *alphaP;
#pragma unroll
    for (int nt = 0; nt < 2; ++nt) {
        int col = nt * 16 + lr;
        float bv = bias[col];
#pragma unroll
        for (int v = 0; v < 8; ++v) {
            int m = lo16 ? v : v + 8;
            int row = m_base + m;
            if (row < n) {
                float val = (nt == 0 ? c0[v] : c1[v]) + bv;
                if (res) val += (float)res[(size_t)row * CH + col];
                val = val >= 0.f ? val : alpha * val;
                Fout[(size_t)row * CH + col] = (_Float16)val;
            }
        }
    }
}

// ---------------------------------------------------------------------------
// Head: logits = F @ lin_W + lin_b (32->255, WMMA over 16 padded N-tiles),
// log-softmax CE split across both waves, block reduce, 1 atomic per WG.
// Block = 64 thr = 2 waves = 32 points.
// ---------------------------------------------------------------------------
__global__ void head_kernel(const _Float16* __restrict__ F,      // (N,32) f16
                            const float* __restrict__ linW,      // (32,255)
                            const float* __restrict__ linB,      // (255)
                            const float* __restrict__ curbin,    // (N,8)
                            const int*   __restrict__ bidx,      // (N)
                            const int*   __restrict__ pnum,      // (B)
                            const int*   __restrict__ kern,      // (8)
                            float* __restrict__ out, int n) {
    __shared__ v16h WlV[256 * CH / 16];           // [col][cin] f16, 16KB
    __shared__ float bl[256];
    __shared__ float Lg[32][257];                 // padded: conflict-free
    __shared__ float pmax[2][32];
    __shared__ float psum[2][32];
    _Float16* Wl = (_Float16*)WlV;

    for (int i = threadIdx.x; i < 256 * CH; i += 64) {
        int col = i >> 5, ci = i & 31;
        Wl[i] = (_Float16)((col < 255) ? linW[ci * 255 + col] : 0.f);
    }
    for (int i = threadIdx.x; i < 256; i += 64) bl[i] = (i < 255) ? linB[i] : 0.f;
    __syncthreads();

    const int lane  = threadIdx.x & 31;
    const int wave  = threadIdx.x >> 5;
    const int lr    = lane & 15;
    const bool lo16 = lane < 16;
    const int kbA   = lo16 ? 0 : 16;
    const int kbB   = lo16 ? 0 : 16;
    const int m_base = blockIdx.x * 32 + wave * 16;

    int row  = m_base + lr;
    int rowc = row < n ? row : n - 1;
    const _Float16* p = F + (size_t)rowc * CH + kbA;
    v8h alo = *(const v8h*)(p);
    v8h ahi = *(const v8h*)(p + 8);
    v16h a;
#pragma unroll
    for (int e = 0; e < 8; ++e) { a[e] = alo[e]; a[e + 8] = ahi[e]; }

#pragma unroll
    for (int nt = 0; nt < 16; ++nt) {
        v16h b = *(const v16h*)(Wl + ((nt * 16 + lr) * CH + kbB));
        v8f z = {0.f,0.f,0.f,0.f,0.f,0.f,0.f,0.f};
        v8f c = __builtin_amdgcn_wmma_f32_16x16x32_f16(false, a, false, b,
                                                       (short)0, z, false, false);
        int col = nt * 16 + lr;
        float bv = bl[col];
#pragma unroll
        for (int v = 0; v < 8; ++v) {
            int m = lo16 ? v : v + 8;
            Lg[wave * 16 + m][col] = c[v] + bv;
        }
    }
    __syncthreads();

    // Split log-softmax: thread t handles point (t&31), half (t>>5) of 255.
    {
        int t  = threadIdx.x;
        int pl = t & 31;
        int hf = t >> 5;
        const float* rowL = Lg[pl];
        int j0 = hf * 128;
        int j1 = hf ? 255 : 128;

        float mx = -3.402823466e38f;
        for (int j = j0; j < j1; ++j) mx = fmaxf(mx, rowL[j]);
        pmax[hf][pl] = mx;
        __syncthreads();
        float gmx = fmaxf(pmax[0][pl], pmax[1][pl]);

        float s = 0.f;
        for (int j = j0; j < j1; ++j) s += __expf(rowL[j] - gmx);
        psum[hf][pl] = s;
        __syncthreads();

        if (t < 32) {
            int pt = blockIdx.x * 32 + t;
            bool valid = pt < n;
            int pc = valid ? pt : n - 1;
            float lse = gmx + __logf(psum[0][t] + psum[1][t]);

            int oct = -1;
#pragma unroll
            for (int kk = 0; kk < 8; ++kk)
                oct += ((int)curbin[(size_t)pc * 8 + kk]) << kern[kk];
            oct = oct < 0 ? 0 : (oct > 254 ? 254 : oct);

            float ce = lse - Lg[t][oct];
            float sc = (float)pnum[bidx[pc]];
            float contrib = valid ? ce / sc * (1.4426950408889634f * 0.25f) : 0.f;

#pragma unroll
            for (int off = 16; off > 0; off >>= 1)
                contrib += __shfl_xor(contrib, off, 32);
            if (t == 0) atomicAdd(out, contrib);
        }
    }
}

// ---------------------------------------------------------------------------
extern "C" void kernel_launch(void* const* d_in, const int* in_sizes, int n_in,
                              void* d_out, int out_size, void* d_ws, size_t ws_size,
                              hipStream_t stream) {
    // setup_inputs() order; 'params' pytree leaves sorted by key (JAX order):
    // dec{W1,W2,a1,a2,b1,b2}, init_W, init_b, lin_W, lin_b, pred{W1,W2,a1,a2,b1,b2}
    const float* recF  = (const float*)d_in[0];
    const float* cbin  = (const float*)d_in[1];
    const float* decW1 = (const float*)d_in[2];
    const float* decW2 = (const float*)d_in[3];
    const float* decA1 = (const float*)d_in[4];
    const float* decA2 = (const float*)d_in[5];
    const float* decB1 = (const float*)d_in[6];
    const float* decB2 = (const float*)d_in[7];
    const float* initW = (const float*)d_in[8];
    const float* initB = (const float*)d_in[9];
    const float* linW  = (const float*)d_in[10];
    const float* linB  = (const float*)d_in[11];
    const float* preW1 = (const float*)d_in[12];
    const float* preW2 = (const float*)d_in[13];
    const float* preA1 = (const float*)d_in[14];
    const float* preA2 = (const float*)d_in[15];
    const float* preB1 = (const float*)d_in[16];
    const float* preB2 = (const float*)d_in[17];
    const int*   bidx  = (const int*)d_in[18];
    const int*   nbr   = (const int*)d_in[19];
    const int*   pnum  = (const int*)d_in[20];
    const int*   kern  = (const int*)d_in[21];
    const int n = in_sizes[0];
    float* out = (float*)d_out;

    size_t fbytes = (((size_t)n * CH * sizeof(_Float16)) + 255) & ~(size_t)255;
    _Float16* bufA = (_Float16*)d_ws;
    _Float16* bufB = (_Float16*)((char*)d_ws + fbytes);
    _Float16* bufC = (_Float16*)((char*)d_ws + 2 * fbytes);

    hipMemsetAsync(d_out, 0, sizeof(float), stream);

    int gInit = (n + 255) / 256;
    init_conv_kernel<<<gInit, 256, 0, stream>>>(recF, initW, initB, nbr, bufA, n);

    int gConv = (n + 127) / 128;
    // dec block
    sparse_conv_wmma<<<gConv, 256, 0, stream>>>(bufA, decW1, decB1, decA1, nullptr, nbr, bufB, n);
    sparse_conv_wmma<<<gConv, 256, 0, stream>>>(bufB, decW2, decB2, decA2, bufA,    nbr, bufC, n);
    // pred block
    sparse_conv_wmma<<<gConv, 256, 0, stream>>>(bufC, preW1, preB1, preA1, nullptr, nbr, bufB, n);
    sparse_conv_wmma<<<gConv, 256, 0, stream>>>(bufB, preW2, preB2, preA2, bufC,    nbr, bufA, n);

    int gFin = (n + 31) / 32;
    head_kernel<<<gFin, 64, 0, stream>>>(bufA, linW, linB, cbin, bidx, pnum, kern, out, n);
}
